// FixedRevRNN_24215025615292
// MI455X (gfx1250) — matrix operand order repaired and checked by reading, not compiled, not run on hardware
//
#include <hip/hip_runtime.h>
#include <stdint.h>

// ---------------------------------------------------------------------------
// FixedRevRNN on MI455X (gfx1250, wave32, WMMA).
//   B=32 S=512 DELAY=8 H=64 D=2 V=256 OUT=256
// K1: QR-orthogonalize 12 (64x64) weights, store bf16 (linear).
// K2: persistent scan, 1 WG / batch element, all state in LDS, bf16 WMMA
//     GEMMs; weights pre-swizzled into WMMA B-fragment layout in LDS
//     (2x ds_load_b128 per fragment); TDM double-buffered embedding fetch
//     with pad_interval/pad_amount giving LDS row stride 68 floats.
// K3: swizzle out_linear to WMMA-B bf16 layout (one shot).
// K4: big WMMA GEMM (16384x8192)@(8192x256), contiguous 16B loads per lane.
// Workspace: [12*4096 bf16 W][32*512*8192 bf16 staged][8192*256 bf16 OLswz]
//            ~260.5 MB.
// ---------------------------------------------------------------------------

typedef __attribute__((ext_vector_type(16))) __bf16   bf16x16;
typedef __attribute__((ext_vector_type(8)))  __bf16   bf16x8;
typedef __attribute__((ext_vector_type(2)))  __bf16   bf16x2;
typedef __attribute__((ext_vector_type(8)))  float    f32x8;
typedef __attribute__((ext_vector_type(4)))  float    f32x4;
typedef __attribute__((ext_vector_type(4)))  uint32_t u32x4;
typedef __attribute__((ext_vector_type(8)))  int      i32x8;
typedef __attribute__((ext_vector_type(4)))  int      i32x4;

#define LDM   68          // padded f32 LDS row stride (64 + 4)
#define MATF  (64 * LDM)  // floats per padded 64x64 matrix
#define NT    256         // threads per block (8 waves)

// ---------------- WMMA tile loaders (contiguous k = 16*(lane/16) + e, A and B alike) -------------------

// A tile 16x32 bf16 from padded f32 LDS. lane<16: e0..15 -> K+0..15 ;
// lane>=16: +16.
__device__ __forceinline__ bf16x16 ldA_f32(const float* A, int row0, int k0, int lane) {
  bf16x16 a;
  int m  = row0 + (lane & 15);
  int kb = k0 + ((lane >> 4) << 4);
  const f32x4* p = (const f32x4*)(A + m * LDM + kb);
  f32x4 q0 = p[0], q1 = p[1], q2 = p[2], q3 = p[3];
#pragma unroll
  for (int e = 0; e < 4; ++e) {
    a[e] = (__bf16)q0[e];      a[e + 4]  = (__bf16)q1[e];
    a[e + 8] = (__bf16)q2[e];  a[e + 12] = (__bf16)q3[e];
  }
  return a;
}
// B tile 32x16 from padded f32 LDS (computed matrices as B operand).
__device__ __forceinline__ bf16x16 ldB_f32(const float* Bm, int k0, int col0, int lane) {
  bf16x16 b;
  int n  = col0 + (lane & 15);
  int kb = k0 + ((lane >> 4) << 4);
#pragma unroll
  for (int e = 0; e < 16; ++e) b[e] = (__bf16)Bm[(kb + e) * LDM + n];
  return b;
}
// B fragment from WMMA-swizzled bf16 storage: 32 contiguous bytes per lane.
// Fragment index within a 64x64 weight: ((kc*4 + ct)*32 + lane)*16.
__device__ __forceinline__ bf16x16 ldB_swz(const __bf16* Wsw, int kc, int ct, int lane) {
  const bf16x8* p = (const bf16x8*)(Wsw + ((((kc << 2) + ct) << 5) + lane) * 16);
  bf16x8 lo = p[0], hi = p[1];
  bf16x16 b;
#pragma unroll
  for (int e = 0; e < 8; ++e) { b[e] = lo[e]; b[e + 8] = hi[e]; }
  return b;
}

// D = A(64x64 f32 lds) * W(64x64 swizzled bf16 lds). 8 waves, 2 tiles each.
__device__ void gemm64_w(float* D, const float* A, const __bf16* Wsw, int tid) {
  const int lane = tid & 31, wid = tid >> 5;
#pragma unroll
  for (int s = 0; s < 2; ++s) {
    int tile = wid + (s << 3);
    int row0 = (tile >> 2) << 4, ct = tile & 3;
    f32x8 acc = {};
#pragma unroll
    for (int kc = 0; kc < 2; ++kc) {
      bf16x16 a = ldA_f32(A, row0, kc * 32, lane);
      bf16x16 b = ldB_swz(Wsw, kc, ct, lane);
      acc = __builtin_amdgcn_wmma_f32_16x16x32_bf16(false, a, false, b,
                                                    (short)0, acc, false, false);
    }
#pragma unroll
    for (int r = 0; r < 8; ++r)
      D[(row0 + r + ((lane >> 4) << 3)) * LDM + (ct << 4) + (lane & 15)] = acc[r];
  }
}
// D = A(64x64 f32 lds) * B(64x64 f32 lds).
__device__ void gemm64_f(float* D, const float* A, const float* Bm, int tid) {
  const int lane = tid & 31, wid = tid >> 5;
#pragma unroll
  for (int s = 0; s < 2; ++s) {
    int tile = wid + (s << 3);
    int row0 = (tile >> 2) << 4, col0 = (tile & 3) << 4;
    f32x8 acc = {};
#pragma unroll
    for (int kc = 0; kc < 2; ++kc) {
      bf16x16 a = ldA_f32(A, row0, kc * 32, lane);
      bf16x16 b = ldB_f32(Bm, kc * 32, col0, lane);
      acc = __builtin_amdgcn_wmma_f32_16x16x32_bf16(false, a, false, b,
                                                    (short)0, acc, false, false);
    }
#pragma unroll
    for (int r = 0; r < 8; ++r)
      D[(row0 + r + ((lane >> 4) << 3)) * LDM + col0 + (lane & 15)] = acc[r];
  }
}

// ---------------- elementwise helpers --------------------------------------

__device__ void copy_mat(float* D, const float* S, int tid) {
  for (int i = tid; i < 4096; i += NT) {
    int o = (i >> 6) * LDM + (i & 63);
    D[o] = S[o];
  }
}
__device__ void copy2_mat(float* D1, float* D2, const float* S, int tid) {
  for (int i = tid; i < 4096; i += NT) {
    int o = (i >> 6) * LDM + (i & 63);
    float v = S[o];
    D1[o] = v; D2[o] = v;
  }
}

// dst = relu(instancenorm(src))  (per-row over 64 cols). 4 threads/row.
__device__ void inorm_relu(float* D, const float* S, int tid) {
  int row = tid >> 2, c0 = (tid & 3) << 4;
  float v[16], s = 0.f, ss = 0.f;
#pragma unroll
  for (int i = 0; i < 16; ++i) { float x = S[row * LDM + c0 + i]; v[i] = x; s += x; ss += x * x; }
  s += __shfl_xor(s, 1, 32); s += __shfl_xor(s, 2, 32);
  ss += __shfl_xor(ss, 1, 32); ss += __shfl_xor(ss, 2, 32);
  float mean = s * (1.f / 64.f);
  float inv  = rsqrtf(ss * (1.f / 64.f) - mean * mean + 1e-5f);
#pragma unroll
  for (int i = 0; i < 16; ++i) { float t = (v[i] - mean) * inv; D[row * LDM + c0 + i] = t > 0.f ? t : 0.f; }
}
// dst = relu(instancenorm(p*q))
__device__ void inorm_relu_mul(float* D, const float* P, const float* Q, int tid) {
  int row = tid >> 2, c0 = (tid & 3) << 4;
  float v[16], s = 0.f, ss = 0.f;
#pragma unroll
  for (int i = 0; i < 16; ++i) {
    float x = P[row * LDM + c0 + i] * Q[row * LDM + c0 + i];
    v[i] = x; s += x; ss += x * x;
  }
  s += __shfl_xor(s, 1, 32); s += __shfl_xor(s, 2, 32);
  ss += __shfl_xor(ss, 1, 32); ss += __shfl_xor(ss, 2, 32);
  float mean = s * (1.f / 64.f);
  float inv  = rsqrtf(ss * (1.f / 64.f) - mean * mean + 1e-5f);
#pragma unroll
  for (int i = 0; i < 16; ++i) { float t = (v[i] - mean) * inv; D[row * LDM + c0 + i] = t > 0.f ? t : 0.f; }
}

// In-place modified Gram-Schmidt (Q factor) on 64x64 padded LDS matrix.
__device__ void mgs_qr(float* A, float* scr, int tid) {
  const int lane = tid & 31, wid = tid >> 5;
  for (int j = 0; j < 64; ++j) {
    if (wid == 0) {                       // column norm by wave 0
      float x0 = A[lane * LDM + j], x1 = A[(lane + 32) * LDM + j];
      float s = x0 * x0 + x1 * x1;
      s += __shfl_xor(s, 16, 32); s += __shfl_xor(s, 8, 32);
      s += __shfl_xor(s, 4, 32);  s += __shfl_xor(s, 2, 32); s += __shfl_xor(s, 1, 32);
      if (lane == 0) scr[0] = rsqrtf(s + 1e-20f);
    }
    __syncthreads();
    float inv = scr[0];
    if (tid < 64) A[tid * LDM + j] *= inv;
    __syncthreads();
    int k = j + 1 + (tid >> 2), part = tid & 3;
    if (k < 64) {                         // eliminate: 4 threads per column
      float d = 0.f; int r0 = part << 4;
#pragma unroll
      for (int r = 0; r < 16; ++r) d += A[(r0 + r) * LDM + j] * A[(r0 + r) * LDM + k];
      d += __shfl_xor(d, 1, 32); d += __shfl_xor(d, 2, 32);
#pragma unroll
      for (int r = 0; r < 16; ++r) A[(r0 + r) * LDM + k] -= d * A[(r0 + r) * LDM + j];
    }
    __syncthreads();
  }
}

// ---------------- Tensor Data Mover: embedding[tok] -> LDS, stride 68 ------
// 6-arg builtin on this toolchain: (u32x4 g0, i32x8 g1, i32x4 g2, i32x4 g3,
//                                   i32x8 g4, i32 cpol)
__device__ __forceinline__ void issue_tdm(uint32_t lds_off, const float* emb, int tok) {
  unsigned long long ga = (unsigned long long)(uintptr_t)emb + (unsigned long long)tok * 16384ull;
  u32x4 g0; i32x8 g1; i32x4 g2, g3; i32x8 g4;
  g0[0] = 1u;                                  // count=1, no gather
  g0[1] = lds_off;                             // LDS byte offset
  g0[2] = (uint32_t)ga;                        // global addr lo
  g0[3] = (uint32_t)((ga >> 32) & 0x01FFFFFFull) | (2u << 30);  // addr hi | type=2
  // data_size=4B | pad_enable | pad_interval=64DW(code5) | pad_amount=4DW(code3)
  g1[0] = (int)0x07520000u;
  g1[1] = (int)(64u << 16);                    // tensor_dim0 = 64
  g1[2] = (int)(16384u << 16);                 // tensor_dim1 = V*64 rows
  g1[3] = (int)(64u << 16);                    // tile_dim0 = 64
  g1[4] = 64;                                  // tile_dim1 = 64
  g1[5] = 64;                                  // tensor_dim0_stride = 64
  g1[6] = 0; g1[7] = 0;
  g2[0] = g2[1] = g2[2] = g2[3] = 0;           // 2D tile: dims 2/3 unused
  g3[0] = g3[1] = g3[2] = g3[3] = 0;
  g4[0] = g4[1] = g4[2] = g4[3] = g4[4] = g4[5] = g4[6] = g4[7] = 0;
  __builtin_amdgcn_tensor_load_to_lds(g0, g1, g2, g3, g4, 0);
}

// ---------------- K1: orthogonalize weights, store bf16 (linear) -----------
__global__ __launch_bounds__(NT) void orth_qr_kernel(const float* __restrict__ lp0,
                                                     const float* __restrict__ lp1,
                                                     __bf16* __restrict__ Wq) {
  __shared__ float M[64 * LDM];
  __shared__ float scr[4];
  const int m = blockIdx.x, tid = threadIdx.x;       // m in [0,12)
  const float* src = (m < 6) ? lp0 : lp1;
  const int d = (m % 6) / 3, i = m % 3;
  const int base = d * 12288 + i * 4096;             // lp[d, i*64+r, c]
  for (int idx = tid; idx < 4096; idx += NT)
    M[(idx >> 6) * LDM + (idx & 63)] = src[base + idx];
  __syncthreads();
  mgs_qr(M, scr, tid);
  __syncthreads();
  for (int idx = tid; idx < 4096; idx += NT)
    Wq[m * 4096 + idx] = (__bf16)M[(idx >> 6) * LDM + (idx & 63)];
}

// ---------------- _calc: 2 layers of norm/gemm/norm/gemm/QR ----------------
__device__ void calc_chain(float*& X, float*& TA, float*& TB, const float* SQ,
                           const __bf16* Wbase, float* scr, int tid) {
#pragma unroll 1
  for (int d = 0; d < 2; ++d) {
    const __bf16* w0 = Wbase + (d * 3 + 0) * 4096;
    const __bf16* w1 = Wbase + (d * 3 + 1) * 4096;
    const __bf16* w2 = Wbase + (d * 3 + 2) * 4096;
    inorm_relu(TA, X, tid);               __syncthreads();
    gemm64_w(TB, TA, w0, tid);                               // b = a @ w0
    gemm64_w(X,  SQ, w1, tid);            __syncthreads();   // c = seq @ w1
    inorm_relu_mul(TA, TB, X, tid);       __syncthreads();   // o = relu(inorm(b*c))
    gemm64_w(TB, TA, w2, tid);            __syncthreads();   // o @ w2
    mgs_qr(TB, scr, tid);                 __syncthreads();   // x = Q(o@w2)
    float* t = X; X = TB; TB = t;                            // result -> X
  }
}

// ---------------- K2: persistent scan, 1 WG per batch element --------------
__global__ __launch_bounds__(NT) void rnn_scan_kernel(const int* __restrict__ tokens,
                                                      const float* __restrict__ emb,
                                                      const float* __restrict__ hs,
                                                      const __bf16* __restrict__ Wg,
                                                      __bf16* __restrict__ stg) {
  extern __shared__ char smem[];
  float*  BUF = (float*)smem;                         // 7 padded f32 matrices
  __bf16* Wl  = (__bf16*)(smem + 7 * MATF * 4);       // 12 * 4096 bf16, swizzled
  float*  scr = (float*)(smem + 7 * MATF * 4 + 12 * 4096 * 2);
  const int b = blockIdx.x, tid = threadIdx.x;

  float* S0 = BUF;
  float* S1 = BUF + MATF;
  float* X  = BUF + 2 * MATF;
  float* TA = BUF + 3 * MATF;
  float* TB = BUF + 4 * MATF;
  float* SEQ[2] = { BUF + 5 * MATF, BUF + 6 * MATF };

  // Swizzle weights into WMMA B-fragment layout:
  //   Wl[w*4096 + ((kc*4+ct)*32 + lane)*16 + e] = Wg[w*4096 + k*64 + n]
  //   with n = ct*16 + (lane&15), k = kc*32 + ((lane>>4)<<4) + e.
  for (int fi = tid; fi < 12 * 256; fi += NT) {
    int lane = fi & 31, ct = (fi >> 5) & 3, kc = (fi >> 7) & 1, w = fi >> 8;
    __bf16* dst = Wl + ((size_t)fi << 4);
    int n  = (ct << 4) + (lane & 15);
    int kb = (kc << 5) + ((lane >> 4) << 4);
#pragma unroll
    for (int e = 0; e < 16; ++e) dst[e] = Wg[w * 4096 + (kb + e) * 64 + n];
  }
  for (int i = tid; i < 4096; i += NT) {
    S0[(i >> 6) * LDM + (i & 63)] = hs[i];
    S1[(i >> 6) * LDM + (i & 63)] = hs[4096 + i];
  }
  __syncthreads();

  // prologue: TDM fetch of embedding for t=0
  if ((tid >> 5) == 0) {
    int tok0 = tokens[b * 512];
    issue_tdm((uint32_t)(uintptr_t)(void*)SEQ[0], emb, tok0);
  }

#pragma unroll 1
  for (int t = 0; t < 520; ++t) {
    __syncthreads();
    if ((tid >> 5) == 0) {                 // wave 0: prefetch t+1, wait for t
      if (t + 1 < 520) {
        int tok = (t + 1 < 512) ? tokens[b * 512 + t + 1] : 0;
        issue_tdm((uint32_t)(uintptr_t)(void*)SEQ[(t + 1) & 1], emb, tok);
        __builtin_amdgcn_s_wait_tensorcnt(1);
      } else {
        __builtin_amdgcn_s_wait_tensorcnt(0);
      }
    }
    __syncthreads();
    const float* SQ = SEQ[t & 1];

    // out1 = inp1 @ _calc(inp0, seq, l0)
    copy_mat(X, S0, tid);                       __syncthreads();
    calc_chain(X, TA, TB, SQ, Wl, scr, tid);
    gemm64_f(TA, S1, X, tid);                   __syncthreads();
    copy2_mat(S1, X, TA, tid);                  __syncthreads();

    // out0 = inp0 @ _calc(out1, seq, l1)
    calc_chain(X, TA, TB, SQ, Wl + 6 * 4096, scr, tid);
    gemm64_f(TA, S0, X, tid);                   __syncthreads();
    copy_mat(S0, TA, tid);                      __syncthreads();

    if (t >= 8) {                               // stage flattened state (bf16 pairs)
      size_t rowbase2 = ((size_t)b * 512 + (size_t)(t - 8)) * 4096;
      bf16x2* sp = (bf16x2*)stg;
      for (int i = tid; i < 4096; i += NT) {
        int r = i >> 5, c = (i & 31) << 1;
        float v0, v1;
        if (r < 64) { v0 = S0[r * LDM + c]; v1 = S0[r * LDM + c + 1]; }
        else        { v0 = S1[(r - 64) * LDM + c]; v1 = S1[(r - 64) * LDM + c + 1]; }
        bf16x2 pkt; pkt[0] = (__bf16)v0; pkt[1] = (__bf16)v1;
        sp[rowbase2 + i] = pkt;
      }
    }
  }
}

// ---------------- K3: swizzle out_linear into WMMA B layout ----------------
// OLs[((kc*16 + nt)*32 + lane)*16 + e] = OL[(kc*32 + ((lane>>4)<<4) + e)*256 + nt*16 + (lane&15)]
__global__ __launch_bounds__(NT) void swz_ol_kernel(const float* __restrict__ OL,
                                                    __bf16* __restrict__ OLs) {
  const int kc = blockIdx.x;                     // 0..255
  for (int f = threadIdx.x; f < 512; f += NT) {  // nt(16) x lane(32)
    int lane = f & 31, nt = f >> 5;
    __bf16* dst = OLs + (((size_t)kc * 16 + nt) * 32 + lane) * 16;
    int n  = (nt << 4) + (lane & 15);
    int kb = (kc << 5) + ((lane >> 4) << 4);
#pragma unroll
    for (int e = 0; e < 16; ++e) dst[e] = (__bf16)OL[(size_t)(kb + e) * 256 + n];
  }
}

// ---------------- K4: out = staged(16384x8192 bf16) @ OLswz ----------------
__global__ __launch_bounds__(NT) void out_proj_kernel(const __bf16* __restrict__ Astg,
                                                      const __bf16* __restrict__ OLs,
                                                      float* __restrict__ out) {
  const int lane = threadIdx.x & 31, wid = threadIdx.x >> 5;
  const int tile_m = blockIdx.x << 6;   // 256 blocks
  const int tile_n = blockIdx.y << 6;   // 4 blocks
#pragma unroll
  for (int s = 0; s < 2; ++s) {
    int tile = wid + (s << 3);
    int row0 = tile_m + ((tile >> 2) << 4);
    int ct   = (tile_n >> 4) + (tile & 3);       // global n-tile index 0..15
    f32x8 acc = {};
    const int m = row0 + (lane & 15);
    const __bf16* arow = Astg + (size_t)m * 8192 + ((lane >> 4) << 4);
#pragma unroll 4
    for (int k0 = 0; k0 < 8192; k0 += 32) {
      const bf16x8* ap = (const bf16x8*)(arow + k0);
      __builtin_prefetch((const void*)(ap + 32), 0, 0);      // stream A ahead
      bf16x8 lo = ap[0], hi = ap[1];
      bf16x16 a;
#pragma unroll
      for (int e = 0; e < 8; ++e) { a[e] = lo[e]; a[e + 8] = hi[e]; }
      const bf16x8* bp = (const bf16x8*)(OLs + (((size_t)(k0 >> 5) * 16 + ct) * 32 + lane) * 16);
      bf16x8 blo = bp[0], bhi = bp[1];
      bf16x16 bb;
#pragma unroll
      for (int e = 0; e < 8; ++e) { bb[e] = blo[e]; bb[e + 8] = bhi[e]; }
      acc = __builtin_amdgcn_wmma_f32_16x16x32_bf16(false, a, false, bb,
                                                    (short)0, acc, false, false);
    }
#pragma unroll
    for (int r = 0; r < 8; ++r)
      out[(size_t)(row0 + r + ((lane >> 4) << 3)) * 256 + (ct << 4) + (lane & 15)] = acc[r];
  }
}

// ---------------------------------------------------------------------------
extern "C" void kernel_launch(void* const* d_in, const int* in_sizes, int n_in,
                              void* d_out, int out_size, void* d_ws, size_t ws_size,
                              hipStream_t stream) {
  const int*   tokens = (const int*)d_in[0];
  const float* lp0    = (const float*)d_in[1];
  const float* lp1    = (const float*)d_in[2];
  const float* emb    = (const float*)d_in[3];
  const float* hs     = (const float*)d_in[4];
  const float* ol     = (const float*)d_in[5];
  float*       out    = (float*)d_out;

  __bf16* wq  = (__bf16*)d_ws;                                   // 96 KB
  __bf16* stg = (__bf16*)((char*)d_ws + 98304);                  // 256 MB
  __bf16* ols = (__bf16*)((char*)d_ws + 98304 + 268435456ull);   // 4 MB

  orth_qr_kernel<<<12, NT, 0, stream>>>(lp0, lp1, wq);
  swz_ol_kernel<<<256, NT, 0, stream>>>(ol, ols);

  const size_t main_lds = 7 * MATF * 4 + 12 * 4096 * 2 + 64;  // ~220 KB (<320 KB/WG)
  rnn_scan_kernel<<<32, NT, main_lds, stream>>>(tokens, emb, hs, wq, stg);

  dim3 g2(256, 4);
  out_proj_kernel<<<g2, NT, 0, stream>>>(stg, ols, out);
}
